// MultiHeadSelfAttention_24507083391475
// MI455X (gfx1250) — matrix operand order repaired and checked by reading, not compiled, not run on hardware
//
#include <hip/hip_runtime.h>
#include <hip/hip_bf16.h>
#include <cstdint>

#define DMODEL 1024
#define NHEAD  16
#define DK     64
#define BATCH  4
#define SEQ    2048
#define MROWS  (BATCH * SEQ)   // 8192

typedef __attribute__((ext_vector_type(16))) __bf16 bf16x16;
typedef __attribute__((ext_vector_type(8)))  float  floatx8;

union FragU { bf16x16 v; uint32_t u[8]; };

// ---- Tensor Data Mover availability (probe-guarded; fallback keeps build green) ----
#if defined(__has_builtin)
# if __has_builtin(__builtin_amdgcn_tensor_load_to_lds) && \
     __has_builtin(__builtin_amdgcn_s_wait_tensorcnt)
#  define USE_TDM 1
# endif
#endif
#ifndef USE_TDM
# define USE_TDM 0
#endif

#if USE_TDM
typedef unsigned int uint32x4 __attribute__((ext_vector_type(4)));
typedef int          int32x8  __attribute__((ext_vector_type(8)));
typedef int          int32x4  __attribute__((ext_vector_type(4)));

// Issue a TDM 2D tile load: tile_d1 x tile_d0 elements of 2 bytes, row stride
// `stride0` elements, from global `gptr` into LDS byte offset `lds_off`.
// D# layout per CDNA5 ISA (group0: count/lds/global/type,
// group1: data_size, dims, tile dims, stride). Groups 2/3 zero (<=2D tensor).
__device__ __forceinline__ void tdm_load_tile_2d(uint32_t lds_off, const void* gptr,
                                                 uint32_t tile_d0, uint32_t tile_d1,
                                                 uint32_t stride0) {
  const uint64_t ga = (uint64_t)(uintptr_t)gptr;
  uint32x4 g0;
  g0.x = 0x1u;                                              // count=1, user descriptor
  g0.y = lds_off;                                           // lds_addr (bytes)
  g0.z = (uint32_t)(ga & 0xffffffffu);                      // global_addr[31:0]
  g0.w = (uint32_t)((ga >> 32) & 0x01ffffffu) | (2u << 30); // global_addr[56:32], type=2
  int32x8 g1;
  g1[0] = (int)(1u << 16);            // workgroup_mask=0, data_size=1 (2 bytes)
  g1[1] = (int)(tile_d0 << 16);       // tensor_dim0[15:0] in bits 31:16
  g1[2] = (int)(tile_d1 << 16);       // tensor_dim0[31:16]=0 | tensor_dim1[15:0]
  g1[3] = (int)(tile_d0 << 16);       // tensor_dim1[31:16]=0 | tile_dim0
  g1[4] = (int)tile_d1;               // tile_dim1 | tile_dim2=0
  g1[5] = (int)stride0;               // tensor_dim0_stride[31:0]
  g1[6] = 0;                          // stride0 hi, stride1 lo
  g1[7] = 0;
  const int32x4 z4 = {0, 0, 0, 0};
#if defined(__clang_major__) && (__clang_major__ >= 23)
  const int32x8 z8 = {0, 0, 0, 0, 0, 0, 0, 0};
  __builtin_amdgcn_tensor_load_to_lds(g0, g1, z4, z4, z8, 0);
#else
  __builtin_amdgcn_tensor_load_to_lds(g0, g1, z4, z4, 0);
#endif
}
#endif  // USE_TDM

__device__ __forceinline__ floatx8 wmma_bf16(bf16x16 a, bf16x16 b, floatx8 c) {
  // D = A(16x32 bf16) x B(32x16 bf16) + C(16x16 f32)
  return __builtin_amdgcn_wmma_f32_16x16x32_bf16(false, a, false, b, (short)0, c,
                                                 false, false);
}

__device__ __forceinline__ floatx8 fzero() {
  floatx8 z;
#pragma unroll
  for (int i = 0; i < 8; ++i) z[i] = 0.f;
  return z;
}

// A fragment: 16x32 bf16 tile from row-major [rows][k] source (base at tile origin).
// ISA layout: lane m=lane&15, half=lane>>4; reg r holds K pair k=(r/4)*16+half*8+(r%4)*2.
__device__ __forceinline__ bf16x16 load_a_frag(const uint16_t* base, int lda, int lane) {
  const int half = lane >> 4, lm = lane & 15;
  FragU f;
#pragma unroll
  for (int r = 0; r < 8; ++r) {
    const int k = ((r >> 2) << 4) + (half << 3) + ((r & 3) << 1);
    f.u[r] = *reinterpret_cast<const uint32_t*>(base + (size_t)lm * lda + k);
  }
  return f.v;
}

// B fragment: 32x16 bf16 (KxN) sourced from row-major [n][k] array (base at tile origin).
// ISA layout: lane n=lane&15, half=lane>>4; reg r holds K pair k=(r/4)*16+half*8+(r%4)*2.
__device__ __forceinline__ bf16x16 load_b_frag(const uint16_t* base, int ldb, int lane) {
  const int half = lane >> 4, lm = lane & 15;
  FragU f;
#pragma unroll
  for (int r = 0; r < 8; ++r) {
    const int k = ((r >> 2) << 4) + (half << 3) + ((r & 3) << 1);
    f.u[r] = *reinterpret_cast<const uint32_t*>(base + (size_t)lm * ldb + k);
  }
  return f.v;
}

// ---------------- f32 -> bf16 conversion ----------------
__global__ void cvt_bf16_kernel(const float* __restrict__ src,
                                __hip_bfloat16* __restrict__ dst, int n) {
  for (int i = blockIdx.x * blockDim.x + threadIdx.x; i < n;
       i += gridDim.x * blockDim.x)
    dst[i] = __float2bfloat16(src[i]);
}

// ---------------- fused QKV projection: y = x @ W^T, scatter to [B,H,S,dk] ----------------
__global__ __launch_bounds__(32) void qkv_gemm_kernel(
    const uint16_t* __restrict__ xb,  // [MROWS][DMODEL] bf16
    const uint16_t* __restrict__ wqb, const uint16_t* __restrict__ wkb,
    const uint16_t* __restrict__ wvb,
    __hip_bfloat16* __restrict__ q, __hip_bfloat16* __restrict__ kq,
    __hip_bfloat16* __restrict__ vq) {
  const int lane = threadIdx.x & 31;
  const int m0 = blockIdx.x * 32;      // 32 rows per wave
  const int n0 = blockIdx.y * 64;      // 64 cols per wave
  const int which = blockIdx.z;        // 0=Q 1=K 2=V
  const uint16_t* w = (which == 0) ? wqb : (which == 1) ? wkb : wvb;
  __hip_bfloat16* dst = (which == 0) ? q : (which == 1) ? kq : vq;

  floatx8 acc[2][4];
#pragma unroll
  for (int mi = 0; mi < 2; ++mi)
#pragma unroll
    for (int ni = 0; ni < 4; ++ni) acc[mi][ni] = fzero();

  for (int k0 = 0; k0 < DMODEL; k0 += 32) {
    __builtin_prefetch(w + (size_t)n0 * DMODEL + k0 + 512, 0, 0);
    bf16x16 a0 = load_a_frag(xb + (size_t)m0 * DMODEL + k0, DMODEL, lane);
    bf16x16 a1 = load_a_frag(xb + (size_t)(m0 + 16) * DMODEL + k0, DMODEL, lane);
#pragma unroll
    for (int ni = 0; ni < 4; ++ni) {
      bf16x16 bf = load_b_frag(w + (size_t)(n0 + ni * 16) * DMODEL + k0, DMODEL, lane);
      acc[0][ni] = wmma_bf16(a0, bf, acc[0][ni]);
      acc[1][ni] = wmma_bf16(a1, bf, acc[1][ni]);
    }
  }
  // C layout: row = r + 8*(lane>>4), col = lane&15 within each 16x16 tile.
  const int half = lane >> 4, lm = lane & 15;
#pragma unroll
  for (int mi = 0; mi < 2; ++mi)
#pragma unroll
    for (int ni = 0; ni < 4; ++ni)
#pragma unroll
      for (int r = 0; r < 8; ++r) {
        const int row = m0 + mi * 16 + r + (half << 3);
        const int col = n0 + ni * 16 + lm;
        const int b = row >> 11, s = row & (SEQ - 1);
        const int h = col >> 6, d = col & (DK - 1);
        dst[(((size_t)(b * NHEAD + h)) * SEQ + s) * DK + d] =
            __float2bfloat16(acc[mi][ni][r]);
      }
}

// ---------------- causal flash attention per (b,h) ----------------
// grid: (SEQ/64, BATCH*NHEAD); block: 128 (4 waves, 16 q-rows each)
__global__ __launch_bounds__(128) void attn_kernel(
    const uint16_t* __restrict__ q, const uint16_t* __restrict__ k,
    const uint16_t* __restrict__ v, __hip_bfloat16* __restrict__ attn) {
  __shared__ uint16_t Ks[64 * 64];      // [key][d]   (B-frag source for Q.K^T)
  __shared__ uint16_t Vt[64 * 64];      // [d][key]   (B-frag source for P.V)
  __shared__ uint16_t Ps[4][16 * 64];   // per-wave P staging [row][key]
#if USE_TDM
  __shared__ uint16_t Vrow[64 * 64];    // TDM row-major V staging [key][d]
#endif

  const int wave = threadIdx.x >> 5;
  const int lane = threadIdx.x & 31;
  const int half = lane >> 4, lm = lane & 15;
  const int bh = blockIdx.y;
  const uint16_t* qb = q + (size_t)bh * SEQ * DK;
  const uint16_t* kb_ = k + (size_t)bh * SEQ * DK;
  const uint16_t* vb = v + (size_t)bh * SEQ * DK;
  const int q0w = blockIdx.x * 64 + wave * 16;

  // Q fragments are invariant over the key loop.
  bf16x16 aq0 = load_a_frag(qb + (size_t)q0w * DK + 0, DK, lane);
  bf16x16 aq1 = load_a_frag(qb + (size_t)q0w * DK + 32, DK, lane);

  floatx8 out[4];
#pragma unroll
  for (int ni = 0; ni < 4; ++ni) out[ni] = fzero();
  float mrow[8], lrow[8];
#pragma unroll
  for (int r = 0; r < 8; ++r) { mrow[r] = -3.0e38f; lrow[r] = 0.f; }

  for (int kt = 0; kt <= (int)blockIdx.x; ++kt) {
    __syncthreads();  // previous iteration finished reading Ks/Vt
#if USE_TDM
    // Tensor Data Mover: DMA K tile and row-major V tile into LDS (wave 0 issues;
    // EXEC ignored by tensor ops, TENSORcnt tracks completion).
    if (wave == 0) {
      tdm_load_tile_2d((uint32_t)(uintptr_t)&Ks[0],   kb_ + (size_t)kt * 64 * DK,
                       DK, 64, DK);
      tdm_load_tile_2d((uint32_t)(uintptr_t)&Vrow[0], vb + (size_t)kt * 64 * DK,
                       DK, 64, DK);
      __builtin_amdgcn_s_wait_tensorcnt(0);
    }
    __syncthreads();  // Ks/Vrow visible to all waves
    // LDS->LDS transpose: Vt[d][key] = Vrow[key][d]
    for (int i = threadIdx.x; i < 64 * DK; i += 128) {
      const int key = i >> 6, d = i & 63;
      Vt[d * 64 + key] = Vrow[i];
    }
#else
    // Stage K tile (row-major copy, dword granularity).
    {
      const uint32_t* src = reinterpret_cast<const uint32_t*>(kb_ + (size_t)kt * 64 * DK);
      uint32_t* dst = reinterpret_cast<uint32_t*>(Ks);
      for (int i = threadIdx.x; i < 64 * DK / 2; i += 128) dst[i] = src[i];
    }
    // Stage V tile transposed: Vt[d][key] = V[key][d]; b128 global loads.
    {
      const uint16_t* src = vb + (size_t)kt * 64 * DK;
      for (int i = threadIdx.x; i < 512; i += 128) {
        ushort v8[8];
        *reinterpret_cast<uint4*>(v8) =
            *reinterpret_cast<const uint4*>(src + i * 8);
        const int key = i >> 3, d0 = (i & 7) * 8;
#pragma unroll
        for (int j = 0; j < 8; ++j) Vt[(d0 + j) * 64 + key] = v8[j];
      }
    }
#endif
    __syncthreads();  // staged tiles visible

    const bool diag = (kt == (int)blockIdx.x);
#pragma unroll
    for (int t = 0; t < 4; ++t) {       // four 16-key score subtiles
      floatx8 sc = fzero();
      sc = wmma_bf16(aq0, load_b_frag(Ks + t * 16 * DK + 0, DK, lane), sc);
      sc = wmma_bf16(aq1, load_b_frag(Ks + t * 16 * DK + 32, DK, lane), sc);
      const int keycol = kt * 64 + t * 16 + lm;
#pragma unroll
      for (int r = 0; r < 8; ++r) {
        const int row = q0w + r + (half << 3);
        float s = sc[r] * 0.125f;                    // 1/sqrt(dk)
        if (diag && keycol > row) s = -3.0e38f;      // causal mask
        float tm = s;                                 // row max over 16 lanes of half
#pragma unroll
        for (int off = 1; off < 16; off <<= 1) tm = fmaxf(tm, __shfl_xor(tm, off, 32));
        const float mnew = fmaxf(mrow[r], tm);
        const float scale = __expf(mrow[r] - mnew);
        const float p = __expf(s - mnew);
        float psum = p;
#pragma unroll
        for (int off = 1; off < 16; off <<= 1) psum += __shfl_xor(psum, off, 32);
        lrow[r] = lrow[r] * scale + psum;
        mrow[r] = mnew;
#pragma unroll
        for (int ni = 0; ni < 4; ++ni) out[ni][r] *= scale;
        Ps[wave][(r + (half << 3)) * 64 + t * 16 + lm] =
            __bfloat16_as_ushort(__float2bfloat16(p));
      }
    }
    asm volatile("s_wait_dscnt 0" ::: "memory");  // wave-local LDS RAW (C->A relayout)
    // out += P(16x64) x V(64x64)
#pragma unroll
    for (int ks = 0; ks < 2; ++ks) {
      bf16x16 ap = load_a_frag(&Ps[wave][0] + ks * 32, 64, lane);
#pragma unroll
      for (int ni = 0; ni < 4; ++ni)
        out[ni] = wmma_bf16(ap, load_b_frag(Vt + ni * 16 * 64 + ks * 32, 64, lane),
                            out[ni]);
    }
  }

  // Epilogue: normalize and store bf16 to [B,S,D] (row-major for final GEMM).
  const int b = bh >> 4, h = bh & 15;
#pragma unroll
  for (int ni = 0; ni < 4; ++ni)
#pragma unroll
    for (int r = 0; r < 8; ++r) {
      const int s = q0w + r + (half << 3);
      const float o = out[ni][r] / fmaxf(lrow[r], 1e-9f);
      attn[((size_t)b * SEQ + s) * DMODEL + h * DK + ni * 16 + lm] =
          __float2bfloat16(o);
    }
}

// ---------------- output projection: out = attn @ wo^T (f32 result) ----------------
__global__ __launch_bounds__(32) void out_gemm_kernel(
    const uint16_t* __restrict__ ab,   // [MROWS][DMODEL] bf16
    const uint16_t* __restrict__ wob,  // [DMODEL][DMODEL] bf16
    float* __restrict__ out) {
  const int lane = threadIdx.x & 31;
  const int m0 = blockIdx.x * 32;
  const int n0 = blockIdx.y * 64;

  floatx8 acc[2][4];
#pragma unroll
  for (int mi = 0; mi < 2; ++mi)
#pragma unroll
    for (int ni = 0; ni < 4; ++ni) acc[mi][ni] = fzero();

  for (int k0 = 0; k0 < DMODEL; k0 += 32) {
    __builtin_prefetch(wob + (size_t)n0 * DMODEL + k0 + 512, 0, 0);
    bf16x16 a0 = load_a_frag(ab + (size_t)m0 * DMODEL + k0, DMODEL, lane);
    bf16x16 a1 = load_a_frag(ab + (size_t)(m0 + 16) * DMODEL + k0, DMODEL, lane);
#pragma unroll
    for (int ni = 0; ni < 4; ++ni) {
      bf16x16 bf = load_b_frag(wob + (size_t)(n0 + ni * 16) * DMODEL + k0, DMODEL, lane);
      acc[0][ni] = wmma_bf16(a0, bf, acc[0][ni]);
      acc[1][ni] = wmma_bf16(a1, bf, acc[1][ni]);
    }
  }
  const int half = lane >> 4, lm = lane & 15;
#pragma unroll
  for (int mi = 0; mi < 2; ++mi)
#pragma unroll
    for (int ni = 0; ni < 4; ++ni)
#pragma unroll
      for (int r = 0; r < 8; ++r) {
        const int row = m0 + mi * 16 + r + (half << 3);
        const int col = n0 + ni * 16 + lm;
        out[(size_t)row * DMODEL + col] = acc[mi][ni][r];
      }
}

extern "C" void kernel_launch(void* const* d_in, const int* in_sizes, int n_in,
                              void* d_out, int out_size, void* d_ws, size_t ws_size,
                              hipStream_t stream) {
  const float* x  = (const float*)d_in[0];
  const float* wq = (const float*)d_in[1];
  const float* wk = (const float*)d_in[2];
  const float* wv = (const float*)d_in[3];
  const float* wo = (const float*)d_in[4];
  float* out = (float*)d_out;

  char* ws = (char*)d_ws;
  const size_t XB = (size_t)MROWS * DMODEL * 2;     // 16 MB
  const size_t WB = (size_t)DMODEL * DMODEL * 2;    // 2 MB
  __hip_bfloat16* xb  = (__hip_bfloat16*)(ws);
  __hip_bfloat16* wqb = (__hip_bfloat16*)(ws + XB);
  __hip_bfloat16* wkb = (__hip_bfloat16*)(ws + XB + WB);
  __hip_bfloat16* wvb = (__hip_bfloat16*)(ws + XB + 2 * WB);
  __hip_bfloat16* wob = (__hip_bfloat16*)(ws + XB + 3 * WB);
  __hip_bfloat16* qd  = (__hip_bfloat16*)(ws + XB + 4 * WB);            // [B,H,S,dk]
  __hip_bfloat16* kd  = (__hip_bfloat16*)(ws + 2 * XB + 4 * WB);
  __hip_bfloat16* vd  = (__hip_bfloat16*)(ws + 3 * XB + 4 * WB);
  __hip_bfloat16* ad  = (__hip_bfloat16*)(ws + 4 * XB + 4 * WB);        // [B,S,D]

  const int nx = MROWS * DMODEL, nw = DMODEL * DMODEL;
  cvt_bf16_kernel<<<2048, 256, 0, stream>>>(x, xb, nx);
  cvt_bf16_kernel<<<1024, 256, 0, stream>>>(wq, wqb, nw);
  cvt_bf16_kernel<<<1024, 256, 0, stream>>>(wk, wkb, nw);
  cvt_bf16_kernel<<<1024, 256, 0, stream>>>(wv, wvb, nw);
  cvt_bf16_kernel<<<1024, 256, 0, stream>>>(wo, wob, nw);

  dim3 gq(MROWS / 32, DMODEL / 64, 3);
  qkv_gemm_kernel<<<gq, 32, 0, stream>>>((const uint16_t*)xb, (const uint16_t*)wqb,
                                         (const uint16_t*)wkb, (const uint16_t*)wvb,
                                         qd, kd, vd);

  dim3 ga(SEQ / 64, BATCH * NHEAD);
  attn_kernel<<<ga, 128, 0, stream>>>((const uint16_t*)qd, (const uint16_t*)kd,
                                      (const uint16_t*)vd, ad);

  dim3 go(MROWS / 32, DMODEL / 64);
  out_gemm_kernel<<<go, 32, 0, stream>>>((const uint16_t*)ad, (const uint16_t*)wob, out);
}
